// MultiHeadAttention_53377853555265
// MI455X (gfx1250) — matrix operand order repaired and checked by reading, not compiled, not run on hardware
//
#include <hip/hip_runtime.h>
#include <hip/hip_bf16.h>
#include <cstdint>

// ---- problem constants (match reference) ----
#define B_   4
#define S_   2048
#define D_   1024
#define H_   16
#define DK_  64

typedef __bf16 bf16_t;
typedef bf16_t v16bf __attribute__((ext_vector_type(16)));
typedef bf16_t v8bf  __attribute__((ext_vector_type(8)));
typedef float  v8f   __attribute__((ext_vector_type(8)));

// D = A(16x32 bf16) * B(32x16 bf16) + C(16x16 f32), wave32 CDNA5 WMMA
__device__ __forceinline__ v8f wmma_bf16(v16bf a, v16bf b, v8f c) {
  return __builtin_amdgcn_wmma_f32_16x16x32_bf16(false, a, false, b, (short)0, c,
                                                 false, false);
}

// A-fragment (16x32, row-major source, stride ld elems).
// Layout: lane l: m=l&15, h=l>>4; VGPRs hold K = 16h..16h+15 (contiguous,
// matching B) -> two 16B loads per lane.
__device__ __forceinline__ v16bf load_a_frag(const bf16_t* base, int ld) {
  const int lane = threadIdx.x & 31;
  const int m = lane & 15, hh = lane >> 4;
  const bf16_t* p = base + (size_t)m * ld + hh * 16;
  v8bf lo = *(const v8bf*)p;
  v8bf hi = *(const v8bf*)(p + 8);
  v16bf a;
#pragma unroll
  for (int i = 0; i < 8; ++i) { a[i] = lo[i]; a[i + 8] = hi[i]; }
  return a;
}

// B-fragment (32x16): source stored one row per N, contiguous in K (stride ld).
// lane l: n=l&15, h=l>>4; VGPRs hold K = 16h..16h+15 contiguous -> one 32B load.
__device__ __forceinline__ v16bf load_b_frag(const bf16_t* base, int ld) {
  const int lane = threadIdx.x & 31;
  const int n = lane & 15, hh = lane >> 4;
  return *(const v16bf*)(base + (size_t)n * ld + hh * 16);
}

// Async global->LDS 16-byte copy (CDNA5, tracked by ASYNCcnt).
// ldsOff: byte offset of destination inside LDS (generic-address low 32 bits).
__device__ __forceinline__ void async_copy_b128(uint32_t ldsOff, const void* g) {
  asm volatile("global_load_async_to_lds_b128 %0, %1, off"
               :: "v"(ldsOff), "v"((uint64_t)(uintptr_t)g)
               : "memory");
}
__device__ __forceinline__ void wait_asynccnt0() {
  asm volatile("s_wait_asynccnt 0x0" ::: "memory");
}

// =====================================================================
// Kernel 1: Q/K/V projection  X[8192,1024] @ W^T -> bf16 workspace.
// z=0 -> q_ws[B][H][S][DK], z=1 -> k_ws[B][H][S][DK], z=2 -> vT_ws[B][H][DK][S]
// =====================================================================
__global__ __launch_bounds__(256)
void mha_qkv_proj_kernel(const float* __restrict__ query,
                         const float* __restrict__ key_,
                         const float* __restrict__ value,
                         const float* __restrict__ Wq,
                         const float* __restrict__ Wk,
                         const float* __restrict__ Wv,
                         bf16_t* __restrict__ q_ws,
                         bf16_t* __restrict__ k_ws,
                         bf16_t* __restrict__ vT_ws) {
  __shared__ __align__(32) bf16_t As[64 * 40];   // 64 rows x 32 K, pitch 40
  __shared__ __align__(32) bf16_t Bs[64 * 48];   // 64 N    x 32 K, pitch 48
  __shared__ __align__(32) bf16_t Es[64 * 64];   // epilogue tile

  const int z = blockIdx.z;
  const float* X = (z == 0) ? query : (z == 1) ? key_ : value;
  const float* W = (z == 0) ? Wq    : (z == 1) ? Wk   : Wv;

  const int mt = blockIdx.x;          // 128 tiles of 64 rows over B*S
  const int h  = blockIdx.y;          // 16 tiles of 64 cols == head index
  const int t  = threadIdx.x;
  const int wave = t >> 5;
  const int wm = wave & 3;            // 4 row-waves (16 rows each)
  const int wn = wave >> 2;           // 2 col-waves (32 cols each)

  const size_t row0 = (size_t)mt * 64;
  const int    col0 = h * 64;
  const int ldr = t >> 2;             // staging row 0..63
  const int ldc = (t & 3) * 8;        // staging col {0,8,16,24}

  v8f c0 = {}; v8f c1 = {};

  for (int kk = 0; kk < D_; kk += 32) {
    // stage A chunk (f32 -> bf16)
    {
      const float4* pa = (const float4*)(X + (row0 + ldr) * D_ + kk + ldc);
      float4 x0 = pa[0], x1 = pa[1];
      bf16_t* d = As + ldr * 40 + ldc;
      d[0] = (bf16_t)x0.x; d[1] = (bf16_t)x0.y; d[2] = (bf16_t)x0.z; d[3] = (bf16_t)x0.w;
      d[4] = (bf16_t)x1.x; d[5] = (bf16_t)x1.y; d[6] = (bf16_t)x1.z; d[7] = (bf16_t)x1.w;
      const float4* pw = (const float4*)(W + (size_t)(col0 + ldr) * D_ + kk + ldc);
      float4 w0 = pw[0], w1 = pw[1];
      bf16_t* e = Bs + ldr * 48 + ldc;
      e[0] = (bf16_t)w0.x; e[1] = (bf16_t)w0.y; e[2] = (bf16_t)w0.z; e[3] = (bf16_t)w0.w;
      e[4] = (bf16_t)w1.x; e[5] = (bf16_t)w1.y; e[6] = (bf16_t)w1.z; e[7] = (bf16_t)w1.w;
      if (kk + 32 < D_) {
        __builtin_prefetch(X + (row0 + ldr) * D_ + kk + 32 + ldc, 0, 1);
        __builtin_prefetch(W + (size_t)(col0 + ldr) * D_ + kk + 32 + ldc, 0, 1);
      }
    }
    __syncthreads();
    v16bf a  = load_a_frag(As + wm * 16 * 40, 40);
    v16bf b0 = load_b_frag(Bs + (wn * 32 + 0)  * 48, 48);
    v16bf b1 = load_b_frag(Bs + (wn * 32 + 16) * 48, 48);
    c0 = wmma_bf16(a, b0, c0);
    c1 = wmma_bf16(a, b1, c1);
    __syncthreads();
  }

  // epilogue: C fragments (lane n=l&15, m = r + 8*(l>>4)) -> LDS tile
  {
    const int lane = t & 31, n = lane & 15, hh = lane >> 4;
#pragma unroll
    for (int r = 0; r < 8; ++r) {
      const int m = wm * 16 + r + 8 * hh;
      Es[m * 64 + wn * 32 + 0  + n] = (bf16_t)c0[r];
      Es[m * 64 + wn * 32 + 16 + n] = (bf16_t)c1[r];
    }
  }
  __syncthreads();

  if (z < 2) {
    // head-major store: [b][h][s][dk], 32B per thread
    bf16_t* dst_ws = (z == 0) ? q_ws : k_ws;
    const int r = t >> 2, cg = (t & 3) * 16;
    const size_t g = row0 + r;
    const int bb = (int)(g >> 11), ss = (int)(g & (S_ - 1));
    bf16_t* dst = dst_ws + ((size_t)(bb * H_ + h) * S_ + ss) * DK_ + cg;
    *(v16bf*)dst = *(const v16bf*)(Es + r * 64 + cg);
  } else {
    // transposed store for V: [b][h][dk][s]
    const int dk = t >> 2, sg = (t & 3) * 16;
    v16bf v;
#pragma unroll
    for (int i = 0; i < 16; ++i) v[i] = Es[(sg + i) * 64 + dk];
    const size_t g = row0 + sg;
    const int bb = (int)(g >> 11), ss = (int)(g & (S_ - 1));
    bf16_t* dst = vT_ws + ((size_t)(bb * H_ + h) * DK_ + dk) * S_ + ss;
    *(v16bf*)dst = v;
  }
}

// =====================================================================
// Kernel 2: fused attention for one (b, h, 16-row q-tile).
//   QK^T -> scale -> mask -> softmax -> attn write (f32, once) -> P@V
// LDS: sim f32 16x2052 | pb bf16 16x2056 | red 512 f32 | sc 8x16x16 f32
// =====================================================================
#define SIM_PITCH 2052
#define PB_PITCH  2056
#define K2_SMEM_BYTES (16 * SIM_PITCH * 4 + 16 * PB_PITCH * 2 + 512 * 4 + 8 * 256 * 4)

__global__ __launch_bounds__(256)
void mha_attn_kernel(const bf16_t* __restrict__ q_ws,
                     const bf16_t* __restrict__ k_ws,
                     const bf16_t* __restrict__ vT_ws,
                     const int* __restrict__ mask,
                     float* __restrict__ attn_out,
                     bf16_t* __restrict__ comb_ws) {
  extern __shared__ char smem[];
  float*  sim = (float*)smem;
  bf16_t* pb  = (bf16_t*)(smem + 16 * SIM_PITCH * 4);
  float*  red = (float*)(smem + 16 * SIM_PITCH * 4 + 16 * PB_PITCH * 2);
  float*  sc  = red + 512;

  const int qt = blockIdx.x, h = blockIdx.y, b = blockIdx.z;
  const int t = threadIdx.x, wave = t >> 5, lane = t & 31;
  const int n = lane & 15, hh = lane >> 4;

  // ---- phase 1: sim = (Q K^T) * 1/8, masked; each wave owns a 256-col strip
  const bf16_t* qp = q_ws + ((size_t)(b * H_ + h) * S_ + qt * 16) * DK_;
  const v16bf a0 = load_a_frag(qp, DK_);        // K = 0..31
  const v16bf a1 = load_a_frag(qp + 32, DK_);   // K = 32..63

  const size_t mbase = (size_t)b * S_ * S_ + (size_t)(qt * 16) * S_;
  const int cstrip = wave * 256;
  for (int kb = 0; kb < 16; ++kb) {
    const int kr0 = cstrip + kb * 16;
    const bf16_t* kp = k_ws + ((size_t)(b * H_ + h) * S_ + kr0) * DK_;
    v8f c = {};
    c = wmma_bf16(a0, load_b_frag(kp, DK_), c);
    c = wmma_bf16(a1, load_b_frag(kp + 32, DK_), c);
#pragma unroll
    for (int r = 0; r < 8; ++r) {
      const int m = r + 8 * hh;
      const int col = kr0 + n;
      const int mk = mask[mbase + (size_t)m * S_ + col];
      sim[m * SIM_PITCH + col] = mk ? c[r] * 0.125f : -1e9f;
    }
  }
  __syncthreads();

  // ---- phase 2: row softmax (16 threads per row), write attn once
  const int row = t >> 4, sub = t & 15;
  float mx = -3.402823466e38f;
  for (int j = 0; j < 128; ++j)
    mx = fmaxf(mx, sim[row * SIM_PITCH + sub + 16 * j]);
  red[row * 16 + sub] = mx;
  __syncthreads();
  mx = red[row * 16];
#pragma unroll
  for (int i = 1; i < 16; ++i) mx = fmaxf(mx, red[row * 16 + i]);
  float sum = 0.f;
  for (int j = 0; j < 128; ++j) {
    const int idx = row * SIM_PITCH + sub + 16 * j;
    const float e = __expf(sim[idx] - mx);
    sim[idx] = e;
    sum += e;
  }
  red[256 + row * 16 + sub] = sum;
  __syncthreads();
  sum = 0.f;
#pragma unroll
  for (int i = 0; i < 16; ++i) sum += red[256 + row * 16 + i];
  const float inv = 1.0f / sum;
  float* arow = attn_out + ((size_t)(b * H_ + h) * S_ + qt * 16 + row) * S_;
  for (int j = 0; j < 128; ++j) {
    const int col = sub + 16 * j;
    const float p = sim[row * SIM_PITCH + col] * inv;
    arow[col] = p;                       // required output, written exactly once
    pb[row * PB_PITCH + col] = (bf16_t)p;
  }
  __syncthreads();

  // ---- phase 3: scores = P(16x2048) @ V(2048x64); waves: 4 dk-strips x 2 K-halves
  const int strip = wave & 3, khalf = wave >> 2;
  const bf16_t* vp = vT_ws + ((size_t)(b * H_ + h) * DK_ + strip * 16) * S_;
  v8f c = {};
  for (int kb = 0; kb < 32; ++kb) {
    const int k0 = khalf * 1024 + kb * 32;
    v16bf a  = load_a_frag(pb + k0, PB_PITCH);
    v16bf bf = load_b_frag(vp + k0, S_);
    c = wmma_bf16(a, bf, c);
  }
#pragma unroll
  for (int r = 0; r < 8; ++r)
    sc[wave * 256 + (r + 8 * hh) * 16 + n] = c[r];
  __syncthreads();

  // reduce K-halves, store combined [b][s][h*64+dk] as bf16
  {
    const int rr = t >> 4, cc = t & 15;
#pragma unroll
    for (int j = 0; j < 4; ++j) {
      const float v = sc[j * 256 + rr * 16 + cc] + sc[(j + 4) * 256 + rr * 16 + cc];
      comb_ws[(size_t)(b * S_ + qt * 16 + rr) * D_ + h * DK_ + j * 16 + cc] = (bf16_t)v;
    }
  }
}

// =====================================================================
// Kernel 3: output projection  combined(bf16) @ Wo^T -> f32 d_out
// A staging uses CDNA5 async global->LDS copies (pure bf16 copy, ASYNCcnt).
// =====================================================================
__global__ __launch_bounds__(256)
void mha_out_proj_kernel(const bf16_t* __restrict__ comb_ws,
                         const float* __restrict__ Wo,
                         float* __restrict__ out) {
  __shared__ __align__(32) bf16_t As[64 * 40];
  __shared__ __align__(32) bf16_t Bs[64 * 48];
  __shared__ __align__(16) float  Ef[64 * 64];

  const int mt = blockIdx.x, nt = blockIdx.y;
  const int t = threadIdx.x, wave = t >> 5;
  const int wm = wave & 3, wn = wave >> 2;
  const size_t row0 = (size_t)mt * 64;
  const int col0 = nt * 64;
  const int ldr = t >> 2, ldc = (t & 3) * 8;

  // LDS byte offset of this thread's 16B staging slot (generic addr low bits)
  const uint32_t aSlot =
      (uint32_t)(uintptr_t)(void*)(As + ldr * 40 + ldc);

  v8f c0 = {}, c1 = {};
  for (int kk = 0; kk < D_; kk += 32) {
    // A: async 16B global->LDS copy (already bf16, no conversion needed)
    async_copy_b128(aSlot, comb_ws + (row0 + ldr) * D_ + kk + ldc);
    // B: Wo f32 -> bf16 through VGPRs
    {
      const float4* pw = (const float4*)(Wo + (size_t)(col0 + ldr) * D_ + kk + ldc);
      float4 w0 = pw[0], w1 = pw[1];
      bf16_t* e = Bs + ldr * 48 + ldc;
      e[0] = (bf16_t)w0.x; e[1] = (bf16_t)w0.y; e[2] = (bf16_t)w0.z; e[3] = (bf16_t)w0.w;
      e[4] = (bf16_t)w1.x; e[5] = (bf16_t)w1.y; e[6] = (bf16_t)w1.z; e[7] = (bf16_t)w1.w;
      if (kk + 32 < D_)
        __builtin_prefetch(comb_ws + (row0 + ldr) * D_ + kk + 32 + ldc, 0, 1);
    }
    wait_asynccnt0();          // drain async copies before barrier
    __syncthreads();
    v16bf a = load_a_frag(As + wm * 16 * 40, 40);
    c0 = wmma_bf16(a, load_b_frag(Bs + (wn * 32 + 0)  * 48, 48), c0);
    c1 = wmma_bf16(a, load_b_frag(Bs + (wn * 32 + 16) * 48, 48), c1);
    __syncthreads();
  }

  const int lane = t & 31, n = lane & 15, hh = lane >> 4;
#pragma unroll
  for (int r = 0; r < 8; ++r) {
    const int m = wm * 16 + r + 8 * hh;
    Ef[m * 64 + wn * 32 + 0  + n] = c0[r];
    Ef[m * 64 + wn * 32 + 16 + n] = c1[r];
  }
  __syncthreads();
  const int rr = t >> 2, cg = (t & 3) * 16;
  float4* dst = (float4*)(out + (row0 + rr) * D_ + col0 + cg);
  const float4* srcv = (const float4*)(Ef + rr * 64 + cg);
#pragma unroll
  for (int i = 0; i < 4; ++i) dst[i] = srcv[i];
}

// =====================================================================
extern "C" void kernel_launch(void* const* d_in, const int* in_sizes, int n_in,
                              void* d_out, int out_size, void* d_ws, size_t ws_size,
                              hipStream_t stream) {
  (void)in_sizes; (void)n_in; (void)out_size; (void)ws_size;

  const float* query = (const float*)d_in[0];
  const float* key_  = (const float*)d_in[1];
  const float* value = (const float*)d_in[2];
  const int*   mask  = (const int*)d_in[3];
  const float* Wq    = (const float*)d_in[4];
  const float* Wk    = (const float*)d_in[5];
  const float* Wv    = (const float*)d_in[6];
  const float* Wo    = (const float*)d_in[7];

  const size_t headElems = (size_t)B_ * H_ * S_ * DK_;  // 8,388,608
  bf16_t* q_ws  = (bf16_t*)d_ws;
  bf16_t* k_ws  = q_ws + headElems;
  bf16_t* vT_ws = k_ws + headElems;
  bf16_t* comb  = vT_ws + headElems;                    // 64 MB total ws

  float* proj = (float*)d_out;                          // [B,S,D] f32
  float* attn = proj + (size_t)B_ * S_ * D_;            // [B,H,S,S] f32

  (void)hipFuncSetAttribute((const void*)mha_attn_kernel,
                            hipFuncAttributeMaxDynamicSharedMemorySize,
                            K2_SMEM_BYTES);

  dim3 blk(256);
  mha_qkv_proj_kernel<<<dim3(128, 16, 3), blk, 0, stream>>>(
      query, key_, value, Wq, Wk, Wv, q_ws, k_ws, vT_ws);
  mha_attn_kernel<<<dim3(128, 16, 4), blk, K2_SMEM_BYTES, stream>>>(
      q_ws, k_ws, vT_ws, mask, attn, comb);
  mha_out_proj_kernel<<<dim3(128, 16), blk, 0, stream>>>(comb, Wo, proj);
}
